// GraphAttentionBlock_41308995453086
// MI455X (gfx1250) — matrix operand order repaired and checked by reading, not compiled, not run on hardware
//
#include <hip/hip_runtime.h>
#include <hip/hip_bf16.h>

#define IN_DIM 256
#define HC 256
#define HEADS 8
#define ODIM 32
#define NEG_SLOPE 0.2f
#define LN_EPS 1e-5f

typedef __attribute__((ext_vector_type(16))) __bf16 v16bf;
typedef __attribute__((ext_vector_type(8)))  float  v8f;

union V16u {
    unsigned int ui[8];
    uint4        q[2];
    v16bf        v;
};

// ---- bf16 helpers via bit manipulation (round-to-nearest-even) ----
__device__ __forceinline__ unsigned short f2bf(float f) {
    unsigned int u = __float_as_uint(f);
    unsigned int r = u + 0x7FFFu + ((u >> 16) & 1u);
    return (unsigned short)(r >> 16);
}
__device__ __forceinline__ float bf2f(unsigned short b) {
    return __uint_as_float(((unsigned int)b) << 16);
}
__device__ __forceinline__ void splitPack(float w0, float w1,
                                          unsigned int& hi, unsigned int& lo) {
    unsigned short h0 = f2bf(w0), h1 = f2bf(w1);
    unsigned short l0 = f2bf(w0 - bf2f(h0)), l1 = f2bf(w1 - bf2f(h1));
    hi = (unsigned int)h0 | ((unsigned int)h1 << 16);
    lo = (unsigned int)l0 | ((unsigned int)l1 << 16);
}

// float atomic max via int/uint ordering trick (valid with -inf init)
__device__ __forceinline__ void atomicMaxFloat(float* addr, float val) {
    if (val >= 0.0f)
        atomicMax((int*)addr, __float_as_int(val));
    else
        atomicMin((unsigned int*)addr, (unsigned int)__float_as_uint(val));
}

// ---------------------------------------------------------------------------
// Pack W [256,256] f32 (row-major W[k][n]) into WMMA B-fragment bf16 hi/lo.
// Fragment ft = kt*16 + nt. Lane L: n = nt*16+(L&15), e=2d+p: K = (L>>4)*8 + e (e<8), 16 + (L>>4)*8 + (e-8) (e>=8).
// Storage: pack[ft*256 + lane*8 + d].
// ---------------------------------------------------------------------------
__global__ void pack_W_kernel(const float* __restrict__ W,
                              unsigned int* __restrict__ whi,
                              unsigned int* __restrict__ wlo) {
    int idx = blockIdx.x * blockDim.x + threadIdx.x;   // 128*32*8 = 32768
    if (idx >= 128 * 32 * 8) return;
    int dw   = idx & 7;
    int lane = (idx >> 3) & 31;
    int ft   = idx >> 8;
    int kt   = ft >> 4;
    int nt   = ft & 15;
    int n    = nt * 16 + (lane & 15);
    int k0   = (dw < 4) ? (kt * 32 + (lane >> 4) * 8 + dw * 2)
                        : (kt * 32 + 16 + (lane >> 4) * 8 + (dw - 4) * 2);
    splitPack(W[(size_t)k0 * HC + n], W[(size_t)(k0 + 1) * HC + n],
              whi[idx], wlo[idx]);
}

// ---------------------------------------------------------------------------
// Pack x [N,256] f32 into WMMA A-fragment bf16 hi/lo (amortizes the f32->bf16
// split across the 16 column-tile waves that consume each row tile).
// Fragment (mt, kt). Lane L: M = mt*16+(L&15); element e(=2d+p):
//   e<8 : K = kt*32 +      (L>>4)*8 + e
//   e>=8: K = kt*32 + 16 + (L>>4)*8 + (e-8)
// Storage: pack[((mt*8+kt)*32 + lane)*8 + d]   (thread per dword => the two
// source floats are contiguous in x).
// ---------------------------------------------------------------------------
__global__ void pack_X_kernel(const float* __restrict__ x,
                              unsigned int* __restrict__ axhi,
                              unsigned int* __restrict__ axlo,
                              int N, int mtiles) {
    size_t idx = (size_t)blockIdx.x * blockDim.x + threadIdx.x;
    size_t total = (size_t)mtiles * 8 * 32 * 8;
    if (idx >= total) return;
    int dw   = idx & 7;
    int lane = (idx >> 3) & 31;
    int kt   = (idx >> 8) & 7;
    int mt   = idx >> 11;
    int m    = mt * 16 + (lane & 15);
    if (m >= N) m = N - 1;                     // clamp for ragged tail
    int grp  = lane >> 4;
    int k0   = (dw < 4) ? (kt * 32 + grp * 8 + 2 * dw)
                        : (kt * 32 + 16 + grp * 8 + 2 * (dw - 4));
    const float* xp = x + (size_t)m * IN_DIM + k0;
    splitPack(xp[0], xp[1], axhi[idx], axlo[idx]);
}

// ---------------------------------------------------------------------------
// GEMM: h[N,256] = x @ W via bf16 split-WMMA (a_hi*b_hi + a_hi*b_lo +
// a_lo*b_hi). One wave per 16x16 output tile; pure loads + WMMA inner loop.
// ---------------------------------------------------------------------------
__global__ __launch_bounds__(256)
void gemm_wmma_kernel(const unsigned int* __restrict__ axhi,
                      const unsigned int* __restrict__ axlo,
                      const unsigned int* __restrict__ whi,
                      const unsigned int* __restrict__ wlo,
                      float* __restrict__ h, int N) {
    int lane = threadIdx.x & 31;
    int wv   = threadIdx.x >> 5;
    int tile = blockIdx.x * 8 + wv;
    int mtiles = (N + 15) >> 4;
    if (tile >= mtiles * 16) return;           // wave-uniform
    int mt  = tile >> 4;
    int nt  = tile & 15;
    int grp = lane >> 4;

    v8f acc = {0.f, 0.f, 0.f, 0.f, 0.f, 0.f, 0.f, 0.f};

#pragma unroll
    for (int kt = 0; kt < 8; kt++) {
        V16u ahi, alo, bhi, blo;
        const uint4* ap  = (const uint4*)(axhi + ((size_t)(mt * 8 + kt) * 32 + lane) * 8);
        const uint4* apl = (const uint4*)(axlo + ((size_t)(mt * 8 + kt) * 32 + lane) * 8);
        const uint4* bp  = (const uint4*)(whi + ((size_t)(kt * 16 + nt) * 32 + lane) * 8);
        const uint4* bpl = (const uint4*)(wlo + ((size_t)(kt * 16 + nt) * 32 + lane) * 8);
        ahi.q[0] = ap[0];  ahi.q[1] = ap[1];
        alo.q[0] = apl[0]; alo.q[1] = apl[1];
        bhi.q[0] = bp[0];  bhi.q[1] = bp[1];
        blo.q[0] = bpl[0]; blo.q[1] = bpl[1];
        acc = __builtin_amdgcn_wmma_f32_16x16x32_bf16(false, ahi.v, false, bhi.v,
                                                      (short)0, acc, false, false);
        acc = __builtin_amdgcn_wmma_f32_16x16x32_bf16(false, ahi.v, false, blo.v,
                                                      (short)0, acc, false, false);
        acc = __builtin_amdgcn_wmma_f32_16x16x32_bf16(false, alo.v, false, bhi.v,
                                                      (short)0, acc, false, false);
    }

    // C layout: VGPR r -> row M = mt*16 + r + 8*grp, col = nt*16 + (lane&15)
    float* hp = h + ((size_t)mt * 16 + 8 * grp) * HC + nt * 16 + (lane & 15);
    if (mt * 16 + 16 <= N) {                   // wave-uniform fast path
#pragma unroll
        for (int r = 0; r < 8; r++) hp[(size_t)r * HC] = acc[r];
    } else {
#pragma unroll
        for (int r = 0; r < 8; r++)
            if (mt * 16 + 8 * grp + r < N) hp[(size_t)r * HC] = acc[r];
    }
}

// ---------------------------------------------------------------------------
// Per-node attention projections: a_src/a_dst [N,8]
// ---------------------------------------------------------------------------
__global__ void attn_proj_kernel(const float* __restrict__ h,
                                 const float* __restrict__ att_src,
                                 const float* __restrict__ att_dst,
                                 float* __restrict__ asrc,
                                 float* __restrict__ adst, int N) {
    int t = blockIdx.x * blockDim.x + threadIdx.x;
    if (t >= N * HEADS) return;
    int n = t >> 3, hd = t & 7;
    const float* hp = h + (size_t)n * HC + hd * ODIM;
    const float* as = att_src + hd * ODIM;
    const float* ad = att_dst + hd * ODIM;
    float s = 0.f, d = 0.f;
#pragma unroll
    for (int c = 0; c < ODIM; c++) {
        float v = hp[c];
        s += v * as[c];
        d += v * ad[c];
    }
    asrc[t] = s;
    adst[t] = d;
}

// ---------------------------------------------------------------------------
// Zero out accumulator, zero denom, -inf max buffer
// ---------------------------------------------------------------------------
__global__ void init_kernel(float* __restrict__ out, float* __restrict__ mbuf,
                            float* __restrict__ denom, int N) {
    size_t t = (size_t)blockIdx.x * blockDim.x + threadIdx.x;
    if (t < (size_t)N * HC) out[t] = 0.f;
    if (t < (size_t)N * HEADS) {
        mbuf[t]  = -3.402823466e38f;
        denom[t] = 0.f;
    }
}

// ---------------------------------------------------------------------------
// Pass 1: per-(dst,head) max of leaky_relu(a_src[src]+a_dst[dst])
// ---------------------------------------------------------------------------
__global__ void edge_max_kernel(const int* __restrict__ ei,
                                const float* __restrict__ asrc,
                                const float* __restrict__ adst,
                                float* __restrict__ mbuf, int E, int N) {
    int e = blockIdx.x * blockDim.x + threadIdx.x;
    if (e >= E + N) return;
    int s, d;
    if (e < E) { s = ei[e]; d = ei[E + e]; }
    else       { s = d = e - E; }
#pragma unroll
    for (int hd = 0; hd < HEADS; hd++) {
        float l = asrc[s * HEADS + hd] + adst[d * HEADS + hd];
        l = l > 0.f ? l : NEG_SLOPE * l;
        atomicMaxFloat(&mbuf[d * HEADS + hd], l);
    }
}

// ---------------------------------------------------------------------------
// Pass 2: one wave per edge. Accumulates exp-weights into denom and
// unnormalized weighted features into out (segment-sum via atomics).
// lane = feature within head; head = loop index => weight is wave-uniform.
// ---------------------------------------------------------------------------
__global__ __launch_bounds__(256)
void edge_scatter_kernel(const int* __restrict__ ei,
                         const float* __restrict__ h,
                         const float* __restrict__ asrc,
                         const float* __restrict__ adst,
                         const float* __restrict__ mbuf,
                         float* __restrict__ denom,
                         float* __restrict__ out, int E, int N) {
    int wave = blockIdx.x * (blockDim.x >> 5) + (threadIdx.x >> 5);
    int lane = threadIdx.x & 31;
    if (wave >= E + N) return;                  // wave-uniform
    int s, d;
    if (wave < E) { s = ei[wave]; d = ei[E + wave]; }
    else          { s = d = wave - E; }
    const float* hs = h + (size_t)s * HC;
    float* od = out + (size_t)d * HC;
#pragma unroll
    for (int hd = 0; hd < HEADS; hd++) {
        float l = asrc[s * HEADS + hd] + adst[d * HEADS + hd];
        l = l > 0.f ? l : NEG_SLOPE * l;
        float w = expf(l - mbuf[d * HEADS + hd]);
        if (lane == 0) atomicAdd(&denom[d * HEADS + hd], w);
        atomicAdd(&od[hd * ODIM + lane], w * hs[hd * ODIM + lane]);
    }
}

// ---------------------------------------------------------------------------
// Pass 3: normalize by denom, add bias, LayerNorm over 256 features.
// One 256-thread block per node, in place on out.
// ---------------------------------------------------------------------------
__global__ __launch_bounds__(256)
void finalize_kernel(float* __restrict__ out,
                     const float* __restrict__ denom,
                     const float* __restrict__ bias,
                     const float* __restrict__ gamma,
                     const float* __restrict__ beta, int N) {
    int n = blockIdx.x;
    int f = threadIdx.x;
    float v = out[(size_t)n * HC + f] / (denom[n * HEADS + (f >> 5)] + 1e-16f)
              + bias[f];
    __shared__ float sbuf[HC];
    sbuf[f] = v;
    __syncthreads();
    for (int st = 128; st > 0; st >>= 1) {
        if (f < st) sbuf[f] += sbuf[f + st];
        __syncthreads();
    }
    float mu = sbuf[0] * (1.0f / HC);
    __syncthreads();
    float dv = v - mu;
    sbuf[f] = dv * dv;
    __syncthreads();
    for (int st = 128; st > 0; st >>= 1) {
        if (f < st) sbuf[f] += sbuf[f + st];
        __syncthreads();
    }
    float var = sbuf[0] * (1.0f / HC);
    out[(size_t)n * HC + f] = gamma[f] * dv * rsqrtf(var + LN_EPS) + beta[f];
}

// ---------------------------------------------------------------------------
extern "C" void kernel_launch(void* const* d_in, const int* in_sizes, int n_in,
                              void* d_out, int out_size, void* d_ws, size_t ws_size,
                              hipStream_t stream) {
    const float* x        = (const float*)d_in[0];
    const int*   ei       = (const int*)d_in[1];
    const float* W        = (const float*)d_in[2];
    const float* att_src  = (const float*)d_in[3];
    const float* att_dst  = (const float*)d_in[4];
    const float* bias     = (const float*)d_in[5];
    const float* ln_gamma = (const float*)d_in[6];
    const float* ln_beta  = (const float*)d_in[7];
    int N = in_sizes[0] / IN_DIM;
    int E = in_sizes[1] / 2;
    int mtiles = (N + 15) / 16;
    float* out = (float*)d_out;

    char* ws = (char*)d_ws;
    size_t off = 0;
    float* h = (float*)(ws + off);                 off += (size_t)N * HC * sizeof(float);
    unsigned int* whi  = (unsigned int*)(ws + off); off += 32768 * sizeof(unsigned int);
    unsigned int* wlo  = (unsigned int*)(ws + off); off += 32768 * sizeof(unsigned int);
    unsigned int* axhi = (unsigned int*)(ws + off); off += (size_t)mtiles * 2048 * sizeof(unsigned int);
    unsigned int* axlo = (unsigned int*)(ws + off); off += (size_t)mtiles * 2048 * sizeof(unsigned int);
    float* asrc  = (float*)(ws + off);             off += (size_t)N * HEADS * sizeof(float);
    float* adst  = (float*)(ws + off);             off += (size_t)N * HEADS * sizeof(float);
    float* mbuf  = (float*)(ws + off);             off += (size_t)N * HEADS * sizeof(float);
    float* denom = (float*)(ws + off);             off += (size_t)N * HEADS * sizeof(float);

    // 1) pack W and x into bf16 hi/lo WMMA fragments
    pack_W_kernel<<<128, 256, 0, stream>>>(W, whi, wlo);
    size_t xfrag = (size_t)mtiles * 2048;
    pack_X_kernel<<<(unsigned)((xfrag + 255) / 256), 256, 0, stream>>>(
        x, axhi, axlo, N, mtiles);

    // 2) h = x @ W  (pure WMMA inner loop)
    int tiles = mtiles * 16;
    gemm_wmma_kernel<<<(tiles + 7) / 8, 256, 0, stream>>>(axhi, axlo, whi, wlo, h, N);

    // 3) attention projections
    attn_proj_kernel<<<(N * HEADS + 255) / 256, 256, 0, stream>>>(
        h, att_src, att_dst, asrc, adst, N);

    // 4) init accumulators
    init_kernel<<<(unsigned)(((size_t)N * HC + 255) / 256), 256, 0, stream>>>(
        out, mbuf, denom, N);

    // 5) segment max over incoming edges (incl. self-loops)
    edge_max_kernel<<<(E + N + 255) / 256, 256, 0, stream>>>(ei, asrc, adst, mbuf, E, N);

    // 6) weighted scatter (unnormalized softmax numerators + denominators)
    edge_scatter_kernel<<<(E + N + 7) / 8, 256, 0, stream>>>(
        ei, h, asrc, adst, mbuf, denom, out, E, N);

    // 7) normalize + bias + LayerNorm
    finalize_kernel<<<N, 256, 0, stream>>>(out, denom, bias, ln_gamma, ln_beta, N);
}
